// AFT_Full_10746008174627
// MI455X (gfx1250) — matrix operand order repaired and checked by reading, compiled, not run on hardware
//
#include <hip/hip_runtime.h>
#include <hip/hip_bf16.h>

typedef _Float16 half_t;
typedef _Float16 v16h __attribute__((ext_vector_type(16)));
typedef _Float16 v8h  __attribute__((ext_vector_type(8)));
typedef float    v8f  __attribute__((ext_vector_type(8)));

#define BB   8
#define CC   3
#define HH   64
#define WW   64
#define TT   4096      // HH*WW tokens
#define HID  128
#define NTOT 2048      // 8 batches * 256 (num||den)

// ---------------------------------------------------------------------------
// Fragment helpers (layouts per CDNA5 ISA 7.12.2, wave32)
// ---------------------------------------------------------------------------
__device__ __forceinline__ v16h mk16(v8h lo, v8h hi) {
    v16h r;
#pragma unroll
    for (int i = 0; i < 8; ++i) { r[i] = lo[i]; r[i + 8] = hi[i]; }
    return r;
}

// A fragment: 16x32 f16 tile from row-major [.., lda] matrix at (row0, k0).
// lane<16: K = sel*8 + {0..7, 16..23}; lane>=16: same with sel=1.
__device__ __forceinline__ v16h load_a_frag(const half_t* __restrict__ A,
                                            int row0, int k0, int lda, int lane) {
    int m   = lane & 15;
    int sel = lane >> 4;
    const half_t* p = A + (size_t)(row0 + m) * lda + k0 + sel * 8;
    v8h lo = *(const v8h*)(p);
    v8h hi = *(const v8h*)(p + 16);
    return mk16(lo, hi);
}

// B fragment: 32x16 f16 tile; B stored N-major (Bt[n][k], row-major, stride ldb).
// lane<16: K = sel*8 + {0..7, 16..23}; lane>=16: same with sel=1.
__device__ __forceinline__ v16h load_b_frag(const half_t* __restrict__ Bt,
                                            int k0, int n0, int ldb, int lane) {
    int n   = lane & 15;
    int sel = lane >> 4;
    const half_t* p = Bt + (size_t)(n0 + n) * ldb + k0 + sel * 8;
    v8h lo = *(const v8h*)(p);
    v8h hi = *(const v8h*)(p + 16);
    return mk16(lo, hi);
}

// ---------------------------------------------------------------------------
// 1) q/k/v projections: [B,T,3] @ [3,128]  (K=3 -> plain VALU)
// ---------------------------------------------------------------------------
__global__ void k_proj(const float* __restrict__ x,
                       const float* __restrict__ wq, const float* __restrict__ bq,
                       const float* __restrict__ wk, const float* __restrict__ bk,
                       const float* __restrict__ wv, const float* __restrict__ bv,
                       float* __restrict__ q, float* __restrict__ k,
                       float* __restrict__ v) {
    int idx = blockIdx.x * blockDim.x + threadIdx.x;
    if (idx >= BB * TT * HID) return;
    int h  = idx & (HID - 1);
    int bt = idx >> 7;                       // b*T + t
    const float* xp = x + (size_t)bt * 3;    // memory-order reshape [B,T,C]
    float x0 = xp[0], x1 = xp[1], x2 = xp[2];
    q[idx] = x0 * wq[h] + x1 * wq[HID + h] + x2 * wq[2 * HID + h] + bq[h];
    k[idx] = x0 * wk[h] + x1 * wk[HID + h] + x2 * wk[2 * HID + h] + bk[h];
    v[idx] = x0 * wv[h] + x1 * wv[HID + h] + x2 * wv[2 * HID + h] + bv[h];
}

// ---------------------------------------------------------------------------
// 2) max over batch of k; convert u,v rows [:T] to f16
// ---------------------------------------------------------------------------
__global__ void k_maxk_cvt(const float* __restrict__ k,
                           const float* __restrict__ u, const float* __restrict__ v,
                           float* __restrict__ maxk,
                           half_t* __restrict__ u_h, half_t* __restrict__ v_h) {
    int idx = blockIdx.x * blockDim.x + threadIdx.x;
    if (idx >= TT * HID) return;
    float m = k[idx];
#pragma unroll
    for (int b = 1; b < BB; ++b) m = fmaxf(m, k[(size_t)b * TT * HID + idx]);
    maxk[idx] = m;
    u_h[idx] = (half_t)u[idx];
    v_h[idx] = (half_t)v[idx];
}

// ---------------------------------------------------------------------------
// 3) GEMM1: ewb = exp(u_h @ v_h^T), f16 out. M=N=4096, K=128.
//    One wave -> 32x64 tile, 8 WMMAs per 32-wide K step.
// ---------------------------------------------------------------------------
__global__ void k_gemm_wb(const half_t* __restrict__ u_h,
                          const half_t* __restrict__ v_h,
                          half_t* __restrict__ ewb) {
    int wid  = (blockIdx.x * blockDim.x + threadIdx.x) >> 5;
    int lane = threadIdx.x & 31;
    int rg = wid >> 6;        // 128 row groups of 32
    int cg = wid & 63;        // 64 col groups of 64
    int row0 = rg * 32, col0 = cg * 64;

    v8f acc[2][4] = {};
#pragma unroll
    for (int k0 = 0; k0 < HID; k0 += 32) {
        v16h a[2], b[4];
#pragma unroll
        for (int i = 0; i < 2; ++i) a[i] = load_a_frag(u_h, row0 + i * 16, k0, HID, lane);
#pragma unroll
        for (int j = 0; j < 4; ++j) b[j] = load_b_frag(v_h, k0, col0 + j * 16, HID, lane);
#pragma unroll
        for (int i = 0; i < 2; ++i)
#pragma unroll
            for (int j = 0; j < 4; ++j)
                acc[i][j] = __builtin_amdgcn_wmma_f32_16x16x32_f16(
                    false, a[i], false, b[j], (short)0, acc[i][j], false, false);
    }
    int n = lane & 15, mb = (lane >> 4) * 8;
#pragma unroll
    for (int i = 0; i < 2; ++i)
#pragma unroll
        for (int j = 0; j < 4; ++j)
#pragma unroll
            for (int r = 0; r < 8; ++r)
                ewb[(size_t)(row0 + i * 16 + mb + r) * TT + col0 + j * 16 + n] =
                    (half_t)__expf(acc[i][j][r]);
}

// ---------------------------------------------------------------------------
// 4) Build N-major B matrix Bt2[n][j], n = b*256 + (h | 128+h):
//    cols 0..127 -> exp_k*val, 128..255 -> exp_k  (f16)
// ---------------------------------------------------------------------------
__global__ void k_bmat(const float* __restrict__ k, const float* __restrict__ v,
                       const float* __restrict__ maxk, half_t* __restrict__ bt2) {
    int idx = blockIdx.x * blockDim.x + threadIdx.x;
    if (idx >= NTOT * TT) return;
    int j  = idx & (TT - 1);
    int n  = idx >> 12;
    int b  = n >> 8;
    int hh = n & 255;
    int h  = hh & 127;
    size_t base = ((size_t)b * TT + j) * HID + h;
    float e = __expf(k[base] - maxk[(size_t)j * HID + h]);
    float o = (hh < 128) ? e * v[base] : e;
    bt2[idx] = (half_t)o;
}

// ---------------------------------------------------------------------------
// 5) GEMM2 (dominant, 68.7 GFLOP): nd = ewb @ Bt2^T.
//    M=4096, N=2048, K=4096. One wave -> 64x64 tile (max reuse: total L2
//    traffic ~2.1 GB vs 3.2 GB for 32x64). 128 K-steps, 16 WMMAs/step,
//    acc = 128 VGPRs (wave32 budget allows up to 1024).
// ---------------------------------------------------------------------------
__global__ void __launch_bounds__(256)
k_gemm_nd(const half_t* __restrict__ ewb, const half_t* __restrict__ bt2,
          float* __restrict__ nd) {
    int wid  = (blockIdx.x * blockDim.x + threadIdx.x) >> 5;
    int lane = threadIdx.x & 31;
    int rg = wid >> 5;        // 64 row groups of 64
    int cg = wid & 31;        // 32 col groups of 64
    int row0 = rg * 64, col0 = cg * 64;

    v8f acc[4][4] = {};
    for (int k0 = 0; k0 < TT; k0 += 32) {
        v16h a[4], b[4];
#pragma unroll
        for (int i = 0; i < 4; ++i) a[i] = load_a_frag(ewb, row0 + i * 16, k0, TT, lane);
#pragma unroll
        for (int j = 0; j < 4; ++j) b[j] = load_b_frag(bt2, k0, col0 + j * 16, TT, lane);
#pragma unroll
        for (int i = 0; i < 4; ++i)
#pragma unroll
            for (int j = 0; j < 4; ++j)
                acc[i][j] = __builtin_amdgcn_wmma_f32_16x16x32_f16(
                    false, a[i], false, b[j], (short)0, acc[i][j], false, false);
    }
    int n = lane & 15, mb = (lane >> 4) * 8;
#pragma unroll
    for (int i = 0; i < 4; ++i)
#pragma unroll
        for (int j = 0; j < 4; ++j)
#pragma unroll
            for (int r = 0; r < 8; ++r)
                nd[(size_t)(row0 + i * 16 + mb + r) * NTOT + col0 + j * 16 + n] =
                    acc[i][j][r];
}

// ---------------------------------------------------------------------------
// 6) combine: y = sigmoid(q)*num/den, then 128->3 projection. One thread/(b,t).
// ---------------------------------------------------------------------------
__global__ void k_combine(const float* __restrict__ q, const float* __restrict__ nd,
                          const float* __restrict__ ow, const float* __restrict__ ob,
                          float* __restrict__ ybuf) {
    int idx = blockIdx.x * blockDim.x + threadIdx.x;
    if (idx >= BB * TT) return;
    int b = idx >> 12;
    int t = idx & (TT - 1);
    float a0 = 0.f, a1 = 0.f, a2 = 0.f;
    const float* ndrow = nd + (size_t)t * NTOT + b * 256;
    const float* qrow  = q + ((size_t)b * TT + t) * HID;
#pragma unroll 4
    for (int h = 0; h < HID; ++h) {
        float num = ndrow[h];
        float den = ndrow[128 + h];
        float sig = 1.f / (1.f + __expf(-qrow[h]));
        float yy  = sig * num / den;
        a0 += yy * ow[h * 3 + 0];
        a1 += yy * ow[h * 3 + 1];
        a2 += yy * ow[h * 3 + 2];
    }
    float* yp = ybuf + (size_t)idx * 3;
    yp[0] = a0 + ob[0]; yp[1] = a1 + ob[1]; yp[2] = a2 + ob[2];
}

// ---------------------------------------------------------------------------
// 7) conv 3->3 k4 s4 + channels-first LayerNorm (C=3). One thread/(b,oh,ow).
//    ybuf [B,T,3] reinterpreted (memory-order .view) as [B,3,64,64].
// ---------------------------------------------------------------------------
__global__ void k_convln(const float* __restrict__ ybuf,
                         const float* __restrict__ cw, const float* __restrict__ cb,
                         const float* __restrict__ lw, const float* __restrict__ lb,
                         float* __restrict__ out) {
    int idx = blockIdx.x * blockDim.x + threadIdx.x;
    if (idx >= BB * 16 * 16) return;
    int b  = idx >> 8;
    int oh = (idx >> 4) & 15;
    int ow = idx & 15;
    float o[3] = {cb[0], cb[1], cb[2]};
#pragma unroll
    for (int ic = 0; ic < 3; ++ic)
#pragma unroll
        for (int kh = 0; kh < 4; ++kh)
#pragma unroll
            for (int kw = 0; kw < 4; ++kw) {
                int hh = oh * 4 + kh, ww = ow * 4 + kw;
                float vv = ybuf[(size_t)b * 12288 + ic * 4096 + hh * 64 + ww];
#pragma unroll
                for (int oc = 0; oc < 3; ++oc)
                    o[oc] += vv * cw[((oc * 3 + ic) * 4 + kh) * 4 + kw];
            }
    float mu  = (o[0] + o[1] + o[2]) * (1.f / 3.f);
    float d0 = o[0] - mu, d1 = o[1] - mu, d2 = o[2] - mu;
    float var = (d0 * d0 + d1 * d1 + d2 * d2) * (1.f / 3.f);
    float inv = rsqrtf(var + 1e-6f);
#pragma unroll
    for (int oc = 0; oc < 3; ++oc)
        out[(((size_t)b * 3 + oc) * 16 + oh) * 16 + ow] =
            lw[oc] * (o[oc] - mu) * inv + lb[oc];
}

// ---------------------------------------------------------------------------
extern "C" void kernel_launch(void* const* d_in, const int* in_sizes, int n_in,
                              void* d_out, int out_size, void* d_ws, size_t ws_size,
                              hipStream_t stream) {
    const float* x    = (const float*)d_in[0];
    const float* wq_w = (const float*)d_in[1];
    const float* wq_b = (const float*)d_in[2];
    const float* wk_w = (const float*)d_in[3];
    const float* wk_b = (const float*)d_in[4];
    const float* wv_w = (const float*)d_in[5];
    const float* wv_b = (const float*)d_in[6];
    const float* ow   = (const float*)d_in[7];
    const float* obv  = (const float*)d_in[8];
    const float* u    = (const float*)d_in[9];
    const float* v    = (const float*)d_in[10];
    const float* cw   = (const float*)d_in[11];
    const float* cb   = (const float*)d_in[12];
    const float* lw   = (const float*)d_in[13];
    const float* lb   = (const float*)d_in[14];
    float* out = (float*)d_out;

    // workspace carve (~105 MB). num/den (33.5MB) aliases k/v/maxk (35.7MB):
    char* p = (char*)d_ws;
    float*  qbuf = (float*)p;  p += (size_t)BB * TT * HID * 4;   // 16.8 MB
    float*  kbuf = (float*)p;  p += (size_t)BB * TT * HID * 4;   // 16.8 MB
    float*  vbuf = (float*)p;  p += (size_t)BB * TT * HID * 4;   // 16.8 MB
    float*  maxk = (float*)p;  p += (size_t)TT * HID * 4;        //  2.1 MB
    float*  nd   = kbuf;  // alias: k/v/maxk dead after k_bmat
    half_t* u_h  = (half_t*)p; p += (size_t)TT * HID * 2;        //  1.0 MB
    half_t* v_h  = (half_t*)p; p += (size_t)TT * HID * 2;        //  1.0 MB
    half_t* ewb  = (half_t*)p; p += (size_t)TT * TT * 2;         // 33.5 MB
    half_t* bt2  = (half_t*)p; p += (size_t)NTOT * TT * 2;       // 16.8 MB
    float*  ybuf = (float*)p;  p += (size_t)BB * TT * 3 * 4;     //  0.4 MB

    const int TPB = 256;
    k_proj<<<(BB * TT * HID) / TPB, TPB, 0, stream>>>(
        x, wq_w, wq_b, wk_w, wk_b, wv_w, wv_b, qbuf, kbuf, vbuf);
    k_maxk_cvt<<<(TT * HID) / TPB, TPB, 0, stream>>>(kbuf, u, v, maxk, u_h, v_h);
    // GEMM1: 128*64 = 8192 waves -> 1024 blocks
    k_gemm_wb<<<1024, TPB, 0, stream>>>(u_h, v_h, ewb);
    k_bmat<<<(NTOT * TT) / TPB, TPB, 0, stream>>>(kbuf, vbuf, maxk, bt2);
    // GEMM2: 64*32 = 2048 waves -> 256 blocks (64x64 tile per wave)
    k_gemm_nd<<<256, TPB, 0, stream>>>(ewb, bt2, nd);
    k_combine<<<(BB * TT) / TPB, TPB, 0, stream>>>(qbuf, nd, ow, obv, ybuf);
    k_convln<<<(BB * 16 * 16) / TPB, TPB, 0, stream>>>(ybuf, cw, cb, lw, lb, out);
}
